// LFSTVSR_4647154614612
// MI455X (gfx1250) — matrix operand order repaired and checked by reading, not compiled, not run on hardware
//
#include <hip/hip_runtime.h>
#include <hip/hip_bf16.h>

typedef __attribute__((ext_vector_type(16))) _Float16 v16h;
typedef __attribute__((ext_vector_type(8)))  float    v8f;

#define BATCH 2
#define H1 192
#define W1 192

__device__ __forceinline__ float lrelu_f(float v) { return v > 0.f ? v : 0.1f * v; }
__device__ __forceinline__ int imin_i(int a, int b) { return a < b ? a : b; }

// ---------------------------------------------------------------------------
// NCHW f32 -> channels-last f16 (into a concat buffer half). C=64, Ctot=128.
// ---------------------------------------------------------------------------
template <int H, int W>
__global__ void k_nchw_to_cl(const float* __restrict__ src, _Float16* __restrict__ dst,
                             int dCoff) {
  constexpr unsigned total = (unsigned)BATCH * 64u * H * W;
  for (unsigned i = blockIdx.x * blockDim.x + threadIdx.x; i < total;
       i += gridDim.x * blockDim.x) {
    unsigned x = i % W; unsigned t = i / W;
    unsigned y = t % H; t /= H;
    unsigned c = t & 63u; unsigned b = t >> 6;
    dst[(((size_t)b * H + y) * W + x) * 128 + dCoff + c] = (_Float16)src[i];
  }
}

// ---------------------------------------------------------------------------
// bilinear 2x upsample (half-pixel centers), channels-last f16 C=64,
// scaled, into a concat buffer half (Ctot=128).
// ---------------------------------------------------------------------------
template <int sH, int sW>
__global__ void k_upsample2x(const _Float16* __restrict__ src, _Float16* __restrict__ dst,
                             int dCoff, float scale) {
  constexpr int dH = sH * 2, dW = sW * 2;
  constexpr unsigned total = (unsigned)BATCH * dH * dW * 64u;
  for (unsigned i = blockIdx.x * blockDim.x + threadIdx.x; i < total;
       i += gridDim.x * blockDim.x) {
    unsigned c = i & 63u; unsigned p = i >> 6;
    unsigned x = p % dW; unsigned t = p / dW;
    unsigned y = t % dH; unsigned b = t / dH;
    float hs = fminf(fmaxf((y + 0.5f) * 0.5f - 0.5f, 0.f), (float)(sH - 1));
    float ws = fminf(fmaxf((x + 0.5f) * 0.5f - 0.5f, 0.f), (float)(sW - 1));
    int h0 = (int)floorf(hs); int h1 = h0 + 1 < sH ? h0 + 1 : sH - 1;
    int w0 = (int)floorf(ws); int w1 = w0 + 1 < sW ? w0 + 1 : sW - 1;
    float fh = hs - (float)h0, fw = ws - (float)w0;
    size_t rb = (size_t)b * sH;
    float v00 = (float)src[((rb + h0) * sW + w0) * 64 + c];
    float v10 = (float)src[((rb + h1) * sW + w0) * 64 + c];
    float v01 = (float)src[((rb + h0) * sW + w1) * 64 + c];
    float v11 = (float)src[((rb + h1) * sW + w1) * 64 + c];
    float v = (v00 * (1.f - fh) + v10 * fh) * (1.f - fw) +
              (v01 * (1.f - fh) + v11 * fh) * fw;
    dst[(((size_t)b * dH + y) * dW + x) * 128 + dCoff + c] = (_Float16)(v * scale);
  }
}

// ---------------------------------------------------------------------------
// Pack OIHW f32 weights into exact WMMA A-fragment order (f16).
// K ordering = tap*CIN + cin. A 16-bit 16x32 layout as used here (same k order as B):
// lane L (m=L&15, hi=L>>4), element j: K = hi*16+j
// dst index = ((coutTile*KChunks + kchunk)*32 + lane)*16 + j
// ---------------------------------------------------------------------------
template <int CIN>
__global__ void k_pack_w(const float* __restrict__ w, _Float16* __restrict__ dst,
                         int CoutPad, int CoutReal) {
  constexpr int KC = (CIN * 9) / 32;
  unsigned total = (unsigned)(CoutPad >> 4) * KC * 512u;
  for (unsigned i = blockIdx.x * blockDim.x + threadIdx.x; i < total;
       i += gridDim.x * blockDim.x) {
    unsigned j = i & 15u;
    unsigned lane = (i >> 4) & 31u;
    unsigned rest = i >> 9;
    unsigned kc = rest % KC;
    unsigned ct = rest / KC;
    unsigned hi = lane >> 4, m = lane & 15u;
    unsigned Kl = hi * 16u + j;
    unsigned k = kc * 32u + Kl;
    unsigned tap = k / CIN, cin = k - tap * CIN;
    unsigned cout = ct * 16u + m;
    float v = 0.f;
    if ((int)cout < CoutReal) v = w[(size_t)(cout * CIN + cin) * 9 + tap];
    dst[i] = (_Float16)v;
  }
}

// ---------------------------------------------------------------------------
// Deformable im2col: builds V[pix, 576] f16, channel = tap*64 + cin,
// from x (channels-last, first 64 channels of the concat buffer, stride 128)
// and the f32 offset/mask conv output om [pix, 224] (o1|o2|mask, 216 real).
// ---------------------------------------------------------------------------
template <int H, int W>
__global__ void k_dcn_im2col(const _Float16* __restrict__ x,
                             const float* __restrict__ om, _Float16* __restrict__ V) {
  constexpr unsigned total = (unsigned)BATCH * H * W * 72u;  // 8 groups * 9 taps
  for (unsigned i = blockIdx.x * blockDim.x + threadIdx.x; i < total;
       i += gridDim.x * blockDim.x) {
    unsigned k = i % 9u;
    unsigned g = (i / 9u) & 7u;
    unsigned p = i / 72u;
    unsigned xc = p % W; unsigned t = p / W;
    unsigned yc = t % H; unsigned b = t / H;
    const float* omp = om + (size_t)p * 224;
    float oy = omp[g * 9 + k];
    float ox = omp[72 + g * 9 + k];
    float mk = 1.f / (1.f + expf(-omp[144 + g * 9 + k]));
    float py = (float)yc + (float)((int)(k / 3u) - 1) + oy;
    float px = (float)xc + (float)((int)(k % 3u) - 1) + ox;
    float y0f = floorf(py), x0f = floorf(px);
    int y0 = (int)y0f, x0 = (int)x0f;
    float ly = py - y0f, lx = px - x0f;
    float acc[8];
#pragma unroll
    for (int c = 0; c < 8; ++c) acc[c] = 0.f;
    int   yi[4] = { y0, y0, y0 + 1, y0 + 1 };
    int   xi[4] = { x0, x0 + 1, x0, x0 + 1 };
    float wg[4] = { (1.f - ly) * (1.f - lx), (1.f - ly) * lx,
                    ly * (1.f - lx), ly * lx };
#pragma unroll
    for (int q = 0; q < 4; ++q) {
      if (yi[q] >= 0 && yi[q] < H && xi[q] >= 0 && xi[q] < W) {
        const _Float16* sp = x + (((size_t)b * H + yi[q]) * W + xi[q]) * 128 + g * 8;
#pragma unroll
        for (int c = 0; c < 8; ++c) acc[c] += wg[q] * (float)sp[c];
      }
    }
    _Float16* vp = V + (size_t)p * 576 + k * 64 + g * 8;
#pragma unroll
    for (int c = 0; c < 8; ++c) vp[c] = (_Float16)(acc[c] * mk);
  }
}

// ---------------------------------------------------------------------------
// Implicit-GEMM conv via v_wmma_f32_16x16x32_f16 (wave32).
// KT==9: 3x3 pad-1 conv over channels-last input (CIN channels).
// KT==1: pure GEMM over CIN "channels" (DCN einsum, CIN=576).
// One wave: 2 pixel tiles (32 pixels) x up to 4 Cout tiles (blockIdx.y group)
// = 8 WMMAs per K-chunk; A fragments reused across both pixel tiles.
// Weight (A) fragments for the block's 4 Cout tiles are cooperatively staged
// through a double-buffered LDS slab (8 KB): 128 threads load one 32B slice
// each per K-chunk, cutting the redundant per-wave vmem weight traffic 4x and
// serving A reads from the LDS banks while the B (activation) stream owns the
// vector-memory path. One workgroup barrier per K-chunk separates buffers.
// 4 waves/block => 128 pixels/block; all pixel counts divide 128 exactly so
// EXEC stays all-ones around the WMMAs (ISA requirement).
// outMode: 0 = f16 channels-last, 1 = f32 channels-last, 2 = f32 NCHW.
// ---------------------------------------------------------------------------
template <int CIN, int KT, int H, int W>
__global__ void k_conv_wmma(const _Float16* __restrict__ in,
                            const _Float16* __restrict__ wpk,
                            const float* __restrict__ bias,
                            void* __restrict__ outp, int outMode, int outCoff, int CoutTot,
                            int CoutPad, int CoutReal, int act) {
  constexpr int KC = (CIN * KT) / 32;
  __shared__ _Float16 smemA[2][4 * 512];

  const unsigned tid = threadIdx.x;
  const unsigned lane = tid & 31u;
  const unsigned wv = tid >> 5;
  const unsigned hi = lane >> 4;
  const unsigned nlo = lane & 15u;
  const int CoutTiles = CoutPad >> 4;
  const int tile0 = blockIdx.y * 4;
  int nt = CoutTiles - tile0; if (nt > 4) nt = 4;

  unsigned pids[2];
  int xs[2], ys[2], bs[2];
  pids[0] = blockIdx.x * 128u + wv * 32u + nlo;
  pids[1] = pids[0] + 16u;
#pragma unroll
  for (int pt = 0; pt < 2; ++pt) {
    unsigned p = pids[pt];
    xs[pt] = (int)(p % W);
    unsigned t = p / W;
    ys[pt] = (int)(t % H);
    bs[pt] = (int)(t / H);
  }

  // cooperative A staging: thread (wv, lane) owns the 32B fragment slice of
  // Cout tile `wv` for lane `lane` (clamped for the partial 224-ch group).
  const int stTile = imin_i(tile0 + (int)wv, CoutTiles - 1);
  auto stageA = [&](int kc, int buf) {
    v16h a = *(const v16h*)(wpk + ((size_t)stTile * KC + kc) * 512 + lane * 16);
    *(v16h*)(&smemA[buf][wv * 512 + lane * 16]) = a;
    if (kc + 2 < KC)  // stream the chunk after next (global_prefetch_b8)
      __builtin_prefetch(wpk + ((size_t)stTile * KC + (kc + 2)) * 512 + lane * 16, 0, 1);
  };

  v8f acc[2][4] = {};

  stageA(0, 0);
  __syncthreads();

  int tap = 0, c0 = 0;
#pragma unroll 2
  for (int kc = 0; kc < KC; ++kc) {
    const int cur = kc & 1;
    if (kc + 1 < KC) stageA(kc + 1, cur ^ 1);

    const int dy = (KT == 9) ? tap / 3 - 1 : 0;
    const int dx = (KT == 9) ? tap % 3 - 1 : 0;

    // B fragments: 16 consecutive channels starting at c0 + hi*16 at tap pixel.
    // B k order used here (same as A): lanes 0-15 K=0-15, lanes 16-31 K=16-31, elem j -> K=hi*16+j.
    v16h bf[2];
#pragma unroll
    for (int pt = 0; pt < 2; ++pt) {
      int yy = ys[pt] + dy, xx = xs[pt] + dx;
      v16h tv = {};
      if (yy >= 0 && yy < H && xx >= 0 && xx < W)
        tv = *(const v16h*)(in + (((size_t)bs[pt] * H + yy) * W + xx) * CIN + c0 + hi * 16);
      bf[pt] = tv;
    }

#pragma unroll
    for (int tt = 0; tt < 4; ++tt) {
      if (tt < nt) {
        v16h a = *(const v16h*)(&smemA[cur][tt * 512 + lane * 16]);
        acc[0][tt] = __builtin_amdgcn_wmma_f32_16x16x32_f16(
            false, a, false, bf[0], (short)0, acc[0][tt], false, false);
        acc[1][tt] = __builtin_amdgcn_wmma_f32_16x16x32_f16(
            false, a, false, bf[1], (short)0, acc[1][tt], false, false);
      }
    }

    c0 += 32;
    if (c0 == CIN) { c0 = 0; ++tap; }
    __syncthreads();  // buffer generation boundary (uniform for all waves)
  }

  // D layout: VGPR r, lane -> M = r + hi*8, N = nlo (pixel)
#pragma unroll
  for (int pt = 0; pt < 2; ++pt) {
#pragma unroll
    for (int tt = 0; tt < 4; ++tt) {
      if (tt < nt) {
#pragma unroll
        for (int r = 0; r < 8; ++r) {
          int cout = (tile0 + tt) * 16 + (int)hi * 8 + r;
          if (cout < CoutReal) {
            float v = acc[pt][tt][r] + bias[cout];
            if (act) v = lrelu_f(v);
            if (outMode == 0)
              ((_Float16*)outp)[(size_t)pids[pt] * CoutTot + outCoff + cout] = (_Float16)v;
            else if (outMode == 1)
              ((float*)outp)[(size_t)pids[pt] * CoutTot + outCoff + cout] = v;
            else
              ((float*)outp)[(((size_t)bs[pt] * CoutTot + outCoff + cout) * H + ys[pt]) * W +
                             xs[pt]] = v;
          }
        }
      }
    }
  }
}

// ---------------------------------------------------------------------------
// Host orchestration
// ---------------------------------------------------------------------------
struct CSpec { int w_off, b_off, Cin, CoutPad, CoutReal; };

// Param leaves per direction, alphabetical-dict-flatten offsets:
// 0 L1_dcn.b 1 L1_dcn.om_b 2 L1_dcn.om_w 3 L1_dcn.w 4 L1_fc.b 5 L1_fc.w
// 6 L1_oc1.b 7 L1_oc1.w 8 L1_oc2.b 9 L1_oc2.w 10 L1_oc3.b 11 L1_oc3.w
// 12 L2_dcn.b 13 om_b 14 om_w 15 w 16 L2_fc.b 17 w 18 L2_oc1.b 19 w
// 20 L2_oc2.b 21 w 22 L2_oc3.b 23 w 24 L3_dcn.b 25 om_b 26 om_w 27 w
// 28 L3_oc1.b 29 w 30 L3_oc2.b 31 w
static const CSpec g_specs[16] = {
  {29, 28, 128,  64,  64},  // 0  L3_oc1
  {31, 30,  64,  64,  64},  // 1  L3_oc2
  {26, 25,  64, 224, 216},  // 2  L3_om
  {27, 24,  64,  64,  64},  // 3  L3_dcn.w
  {19, 18, 128,  64,  64},  // 4  L2_oc1
  {21, 20, 128,  64,  64},  // 5  L2_oc2
  {23, 22,  64,  64,  64},  // 6  L2_oc3
  {14, 13,  64, 224, 216},  // 7  L2_om
  {15, 12,  64,  64,  64},  // 8  L2_dcn.w
  {17, 16, 128,  64,  64},  // 9  L2_fc
  { 7,  6, 128,  64,  64},  // 10 L1_oc1
  { 9,  8, 128,  64,  64},  // 11 L1_oc2
  {11, 10,  64,  64,  64},  // 12 L1_oc3
  { 2,  1,  64, 224, 216},  // 13 L1_om
  { 3,  0,  64,  64,  64},  // 14 L1_dcn.w
  { 5,  4, 128,  64,  64},  // 15 L1_fc
};

template <int CIN, int KT, int H, int W>
static void conv_t(const _Float16* in, const _Float16* wpk, const float* bias,
                   void* out, int mode, int coff, int ctot, int CoutPad, int CoutReal,
                   int act, hipStream_t s) {
  dim3 grid((unsigned)((BATCH * H * W) / 128), (unsigned)((CoutPad / 16 + 3) / 4));
  k_conv_wmma<CIN, KT, H, W><<<grid, 128, 0, s>>>(in, wpk, bias, out, mode, coff, ctot,
                                                  CoutPad, CoutReal, act);
}

template <int sH, int sW>
static void up_t(const _Float16* src, _Float16* dst, int dCoff, float scale, hipStream_t s) {
  unsigned tot = (unsigned)BATCH * sH * 2 * sW * 2 * 64;
  k_upsample2x<sH, sW><<<(tot + 255) / 256, 256, 0, s>>>(src, dst, dCoff, scale);
}

template <int H, int W>
static void tocl_t(const float* src, _Float16* dst, int coff, hipStream_t s) {
  unsigned tot = (unsigned)BATCH * 64 * H * W;
  k_nchw_to_cl<H, W><<<(tot + 255) / 256, 256, 0, s>>>(src, dst, coff);
}

template <int H, int W>
static void im2col_t(const _Float16* xcat, const float* om, _Float16* V, hipStream_t s) {
  unsigned tot = (unsigned)BATCH * H * W * 72;
  k_dcn_im2col<H, W><<<(tot + 255) / 256, 256, 0, s>>>(xcat, om, V);
}

extern "C" void kernel_launch(void* const* d_in, const int* in_sizes, int n_in,
                              void* d_out, int out_size, void* d_ws, size_t ws_size,
                              hipStream_t stream) {
  (void)in_sizes; (void)n_in; (void)out_size; (void)ws_size;
  const int B = BATCH;

  // ---- workspace bump allocator ----
  size_t cur = 0;
  auto alloc = [&](size_t bytes) -> void* {
    cur = (cur + 255) & ~(size_t)255;
    void* p = (char*)d_ws + cur;
    cur += bytes;
    return p;
  };
  auto allocH = [&](size_t n) { return (_Float16*)alloc(n * sizeof(_Float16)); };

  // packed weights for 16 convs x 2 directions
  size_t packSize[16], packOff[2][16], po = 0;
  for (int i = 0; i < 16; ++i)
    packSize[i] = (size_t)(g_specs[i].CoutPad >> 4) * ((g_specs[i].Cin * 9) >> 5) * 512;
  for (int d = 0; d < 2; ++d)
    for (int i = 0; i < 16; ++i) { packOff[d][i] = po; po += packSize[i]; }
  _Float16* wpack = allocH(po);

  _Float16* cat1  = allocH((size_t)B * H1 * W1 * 128);
  _Float16* cat2  = allocH((size_t)B * 96 * 96 * 128);
  _Float16* cat3  = allocH((size_t)B * 48 * 48 * 128);
  _Float16* catB1 = allocH((size_t)B * H1 * W1 * 128);  // also reused as catC1
  _Float16* catB2 = allocH((size_t)B * 96 * 96 * 128);  // also reused as catC2
  _Float16* tmpX  = allocH((size_t)B * H1 * W1 * 64);
  _Float16* tmpY  = allocH((size_t)B * H1 * W1 * 64);
  _Float16* off3  = allocH((size_t)B * 48 * 48 * 64);
  _Float16* off2  = allocH((size_t)B * 96 * 96 * 64);
  _Float16* fea3  = allocH((size_t)B * 48 * 48 * 64);
  _Float16* fea2  = allocH((size_t)B * 96 * 96 * 64);
  float*    om    = (float*)alloc((size_t)B * H1 * W1 * 224 * sizeof(float));
  _Float16* V     = allocH((size_t)B * H1 * W1 * 576);

  const float* fin[6];
  for (int i = 0; i < 6; ++i) fin[i] = (const float*)d_in[i];

  // ---- pack all weights (deterministic, every call) ----
  for (int d = 0; d < 2; ++d) {
    int base = d ? 38 : 6;
    for (int i = 0; i < 16; ++i) {
      const CSpec& s = g_specs[i];
      unsigned blocks = (unsigned)((packSize[i] + 255) / 256);
      const float* wsrc = (const float*)d_in[base + s.w_off];
      if (s.Cin == 128)
        k_pack_w<128><<<blocks, 256, 0, stream>>>(wsrc, wpack + packOff[d][i],
                                                  s.CoutPad, s.CoutReal);
      else
        k_pack_w<64><<<blocks, 256, 0, stream>>>(wsrc, wpack + packOff[d][i],
                                                 s.CoutPad, s.CoutReal);
    }
  }

  for (int d = 0; d < 2; ++d) {
    // feaA aligned toward feaB; direction 1 swaps the roles
    const float* A1 = fin[d ? 3 : 0]; const float* A2 = fin[d ? 4 : 1]; const float* A3 = fin[d ? 5 : 2];
    const float* P1 = fin[d ? 0 : 3]; const float* P2 = fin[d ? 1 : 4]; const float* P3 = fin[d ? 2 : 5];

    auto wp = [&](int ci) { return wpack + packOff[d][ci]; };
    auto bp = [&](int ci) { return (const float*)d_in[(d ? 38 : 6) + g_specs[ci].b_off]; };
    auto cpad = [&](int ci) { return g_specs[ci].CoutPad; };
    auto creal = [&](int ci) { return g_specs[ci].CoutReal; };

    tocl_t<H1, W1>(A1, cat1, 0, stream);  tocl_t<H1, W1>(P1, cat1, 64, stream);
    tocl_t<96, 96>(A2, cat2, 0, stream);  tocl_t<96, 96>(P2, cat2, 64, stream);
    tocl_t<48, 48>(A3, cat3, 0, stream);  tocl_t<48, 48>(P3, cat3, 64, stream);

    // ---- L3 ----
    conv_t<128, 9, 48, 48>(cat3, wp(0), bp(0), tmpX, 0, 0, 64, cpad(0), creal(0), 1, stream);
    conv_t<64, 9, 48, 48>(tmpX, wp(1), bp(1), off3, 0, 0, 64, cpad(1), creal(1), 1, stream);
    conv_t<64, 9, 48, 48>(off3, wp(2), bp(2), om, 1, 0, 224, cpad(2), creal(2), 0, stream);
    im2col_t<48, 48>(cat3, om, V, stream);
    conv_t<576, 1, 48, 48>(V, wp(3), bp(3), fea3, 0, 0, 64, cpad(3), creal(3), 1, stream);

    // ---- L2 ----
    conv_t<128, 9, 96, 96>(cat2, wp(4), bp(4), catB2, 0, 0, 128, cpad(4), creal(4), 1, stream);
    up_t<48, 48>(off3, catB2, 64, 2.0f, stream);               // up(L3_off)*2 -> catB2[:,64:]
    conv_t<128, 9, 96, 96>(catB2, wp(5), bp(5), tmpX, 0, 0, 64, cpad(5), creal(5), 1, stream);
    conv_t<64, 9, 96, 96>(tmpX, wp(6), bp(6), off2, 0, 0, 64, cpad(6), creal(6), 1, stream);
    conv_t<64, 9, 96, 96>(off2, wp(7), bp(7), om, 1, 0, 224, cpad(7), creal(7), 0, stream);
    im2col_t<96, 96>(cat2, om, V, stream);
    conv_t<576, 1, 96, 96>(V, wp(8), bp(8), catB2, 0, 0, 128, cpad(8), creal(8), 0, stream);
    up_t<48, 48>(fea3, catB2, 64, 1.0f, stream);               // up(L3_fea) -> catC2[:,64:]
    conv_t<128, 9, 96, 96>(catB2, wp(9), bp(9), fea2, 0, 0, 64, cpad(9), creal(9), 1, stream);

    // ---- L1 ----
    conv_t<128, 9, H1, W1>(cat1, wp(10), bp(10), catB1, 0, 0, 128, cpad(10), creal(10), 1, stream);
    up_t<96, 96>(off2, catB1, 64, 2.0f, stream);               // up(L2_off)*2
    conv_t<128, 9, H1, W1>(catB1, wp(11), bp(11), tmpX, 0, 0, 64, cpad(11), creal(11), 1, stream);
    conv_t<64, 9, H1, W1>(tmpX, wp(12), bp(12), tmpY, 0, 0, 64, cpad(12), creal(12), 1, stream);
    conv_t<64, 9, H1, W1>(tmpY, wp(13), bp(13), om, 1, 0, 224, cpad(13), creal(13), 0, stream);
    im2col_t<H1, W1>(cat1, om, V, stream);
    conv_t<576, 1, H1, W1>(V, wp(14), bp(14), catB1, 0, 0, 128, cpad(14), creal(14), 0, stream);
    up_t<96, 96>(fea2, catB1, 64, 1.0f, stream);               // up(L2_fea)
    conv_t<128, 9, H1, W1>(catB1, wp(15), bp(15), d_out, 2, d * 64, 128, cpad(15), creal(15),
                           0, stream);                          // L1_fc -> d_out NCHW
  }
}
